// MultiHeadAttention_9809705304493
// MI455X (gfx1250) — matrix operand order repaired and checked by reading, not compiled, not run on hardware
//
#include <hip/hip_runtime.h>

typedef __attribute__((ext_vector_type(16))) _Float16     v16h;
typedef __attribute__((ext_vector_type(8)))  float        v8f;
typedef __attribute__((ext_vector_type(4)))  unsigned int u32x4;
typedef __attribute__((ext_vector_type(8)))  int          i32x8;
typedef __attribute__((ext_vector_type(4)))  int          i32x4;

#define B_   4
#define L_   2048
#define D_   1024
#define H_   16
#define DPH_ 64
#define M_   (B_*L_)

#if __has_builtin(__builtin_amdgcn_tensor_load_to_lds) && \
    __has_builtin(__builtin_amdgcn_s_wait_tensorcnt)
#define USE_TDM 1
#else
#define USE_TDM 0
#endif

union AFrag { unsigned int u[8]; v16h h; };
union Q4    { uint4 v; _Float16 h[8]; };

__device__ __forceinline__ unsigned lds_off(const void* p) {
  // Generic pointer to LDS: low 32 bits are the wave-relative LDS byte offset.
  return (unsigned)(size_t)p;
}

#if USE_TDM
// 2D TDM load: tile_d1 rows of tile_d0 f16 elements (row stride stride0 elems)
// from global into LDS, rows packed contiguously (tile_d0*2 bytes per row).
__device__ __forceinline__ void tdm_load_2d(unsigned lds_addr, const void* gptr,
                                            unsigned tile_d0, unsigned tile_d1,
                                            unsigned long long stride0) {
  unsigned long long ga = (unsigned long long)gptr;
  u32x4 g0;
  g0[0] = 1u;                                   // count=1, user mode
  g0[1] = lds_addr;                             // lds_addr
  g0[2] = (unsigned)ga;                         // global_addr[31:0]
  g0[3] = (unsigned)((ga >> 32) & 0x1FFFFFFu)   // global_addr[56:32]
          | (2u << 30);                         // type = 2 ("image")
  i32x8 g1;
  g1[0] = (int)(1u << 16);                      // data_size = 1 (2 bytes)
  g1[1] = (int)((tile_d0 & 0xFFFFu) << 16);     // tensor_dim0[15:0]
  g1[2] = (int)(((tile_d0 >> 16) & 0xFFFFu) |   // tensor_dim0[31:16]
                ((tile_d1 & 0xFFFFu) << 16));   // tensor_dim1[15:0]
  g1[3] = (int)(((tile_d1 >> 16) & 0xFFFFu) |   // tensor_dim1[31:16]
                ((tile_d0 & 0xFFFFu) << 16));   // tile_dim0
  g1[4] = (int)(tile_d1 & 0xFFFFu);             // tile_dim1 (tile_dim2 = 0)
  g1[5] = (int)(unsigned)(stride0 & 0xFFFFFFFFu);        // dim0_stride[31:0]
  g1[6] = (int)(unsigned)((stride0 >> 32) & 0xFFFFu);    // dim0_stride[47:32]
  g1[7] = 0;
  i32x4 z4 = {0, 0, 0, 0};
  i32x8 z8 = {0, 0, 0, 0, 0, 0, 0, 0};
  __builtin_amdgcn_tensor_load_to_lds(g0, g1, z4, z4, z8, 0);
}
#endif

__device__ __forceinline__ float hmax16(float v) {
  v = fmaxf(v, __shfl_xor(v, 1));
  v = fmaxf(v, __shfl_xor(v, 2));
  v = fmaxf(v, __shfl_xor(v, 4));
  v = fmaxf(v, __shfl_xor(v, 8));
  return v;
}
__device__ __forceinline__ float hsum16(float v) {
  v += __shfl_xor(v, 1);
  v += __shfl_xor(v, 2);
  v += __shfl_xor(v, 4);
  v += __shfl_xor(v, 8);
  return v;
}

__global__ void cvt_f32_to_f16(const float* __restrict__ src,
                               _Float16* __restrict__ dst, int n) {
  int i = blockIdx.x * blockDim.x + threadIdx.x;
  int stride = gridDim.x * blockDim.x;
  for (; i < n; i += stride) dst[i] = (_Float16)src[i];
}

// ---------------------------------------------------------------------------
// GEMM: Y[m][n] = sum_k A[m][k] * W[n][k]  (X @ W^T), K = 1024.
// MODE 0: A row-major f16, out f16 scattered to [b,h,l,d]; (acc+bias)*scale
// MODE 1: A is context [b,h,l,d] f16 read as [m][h*64+d]; out fp32 [m][n]
// 256 threads = 8 waves; block tile 64(M) x 128(N); wave tile 16x64.
// TDM double-buffered: chunk i+1 DMA overlaps chunk i WMMAs.
// ---------------------------------------------------------------------------
template<int MODE>
__global__ __launch_bounds__(256)
void gemm_f16_wmma(const _Float16* __restrict__ A, const _Float16* __restrict__ Bw,
                   const float* __restrict__ bias, void* __restrict__ outp, float scale) {
  __shared__ _Float16 ldsA[2][64 * 32];    // [row][k]
  __shared__ _Float16 ldsB[2][128 * 32];   // [n][k] (= weight rows)

  const int tid  = threadIdx.x;
  const int wave = tid >> 5, lane = tid & 31;
  const int l15  = lane & 15;
  const int hi   = (lane >= 16) ? 1 : 0;
  const int Mbase = blockIdx.x * 64;
  const int Nbase = blockIdx.y * 128;
  const int wMi = wave >> 1, wNi = wave & 1;

  v8f acc[4] = {};

#if USE_TDM
  auto issue = [&](int kc, int buf) {
    const _Float16* aBase;
    unsigned long long aStride;
    if (MODE == 0) {
      aBase = A + (long)Mbase * D_ + kc;
      aStride = D_;
    } else {
      int bb = Mbase >> 11, l0 = Mbase & 2047, hh = kc >> 6, d0 = kc & 63;
      aBase = A + ((long)(bb * H_ + hh) * L_ + l0) * DPH_ + d0;
      aStride = DPH_;
    }
    tdm_load_2d(lds_off(&ldsA[buf][0]), aBase, 32, 64, aStride);
    tdm_load_2d(lds_off(&ldsB[buf][0]), Bw + (long)Nbase * D_ + kc, 32, 128, D_);
  };
  if (wave == 0) issue(0, 0);
#endif

  for (int kc = 0; kc < D_; kc += 32) {
    const int buf = (kc >> 5) & 1;
#if USE_TDM
    if (wave == 0) __builtin_amdgcn_s_wait_tensorcnt(0);  // buf[cur] complete
    __syncthreads();           // buf[cur] visible; prior reads of buf^1 done
    if (wave == 0 && kc + 32 < D_) issue(kc + 32, buf ^ 1);  // overlaps compute
    if (kc + 64 < D_)   // warm L2 two chunks ahead
      __builtin_prefetch(&Bw[(long)(Nbase + (tid >> 1)) * D_ + kc + 64 + (tid & 1) * 16], 0, 1);
#else
    __syncthreads();
    { // cooperative A tile load: 64 rows x 32 halves
      int row = tid >> 2, k0 = (tid & 3) * 8;
      int m = Mbase + row, k = kc + k0;
      long idx;
      if (MODE == 0) {
        idx = (long)m * D_ + k;
      } else {
        int bb = m >> 11, l = m & 2047, hh = k >> 6, d = k & 63;
        idx = ((long)(bb * H_ + hh) * L_ + l) * DPH_ + d;
      }
      *(uint4*)&ldsA[buf][row * 32 + k0] = *(const uint4*)&A[idx];
    }
    { // cooperative B tile: 128 rows x 32 halves
      #pragma unroll
      for (int i = 0; i < 2; ++i) {
        int idx = tid + i * 256;
        int n = idx >> 2, k0 = (idx & 3) * 8;
        *(uint4*)&ldsB[buf][n * 32 + k0] =
            *(const uint4*)&Bw[(long)(Nbase + n) * D_ + kc + k0];
      }
    }
    __syncthreads();
#endif
    const unsigned int* ldsAu = (const unsigned int*)&ldsA[buf][0];
    const unsigned int* ldsBu = (const unsigned int*)&ldsB[buf][0];

    AFrag a;
    { // A operand 16x32: VGPR v pair K = (v<4?0:16)+(hi?8:0)+2*(v&3)
      int rb = (wMi * 16 + l15) * 16;
      #pragma unroll
      for (int v = 0; v < 8; ++v)
        a.u[v] = ldsAu[rb + (v & 3) + (hi ? 4 : 0) + ((v < 4) ? 0 : 8)];
    }
    #pragma unroll
    for (int t = 0; t < 4; ++t) {
      AFrag bb; // B operand 32x16: VGPR v pair K = (v<4?0:16)+(hi?8:0)+2*(v&3), N = l15
      int nb = (wNi * 64 + t * 16 + l15) * 16;
      #pragma unroll
      for (int v = 0; v < 8; ++v) bb.u[v] = ldsBu[nb + (v & 3) + (hi ? 4 : 0) + ((v < 4) ? 0 : 8)];
      acc[t] = __builtin_amdgcn_wmma_f32_16x16x32_f16(
          false, a.h, false, bb.h, (short)0, acc[t], false, false);
    }
  }

  // Epilogue. C layout: VGPR r -> M = r + (hi?8:0), N = l15 (per 16-col tile)
  #pragma unroll
  for (int t = 0; t < 4; ++t) {
    int nglob = Nbase + wNi * 64 + t * 16 + l15;
    float bv = bias[nglob];
    #pragma unroll
    for (int r = 0; r < 8; ++r) {
      int m = Mbase + wMi * 16 + r + (hi ? 8 : 0);
      float val = (acc[t][r] + bv) * scale;
      if (MODE == 0) {
        int bb = m >> 11, l = m & 2047, hh = nglob >> 6, d = nglob & 63;
        ((_Float16*)outp)[((long)(bb * H_ + hh) * L_ + l) * DPH_ + d] = (_Float16)val;
      } else {
        ((float*)outp)[(long)m * D_ + nglob] = val;
      }
    }
  }
}

// ---------------------------------------------------------------------------
// Flash attention: grid (qblock=16, head=16, batch=4), 256 threads = 8 waves.
// Each wave: 16 q-rows, dph=64, KV streamed in chunks of 32 with online
// softmax. K tile TDM double-buffered; V loaded transposed manually.
// 8 WMMAs per wave per chunk (4 QK^T, 4 P@V).
// ---------------------------------------------------------------------------
__global__ __launch_bounds__(256)
void flash_attn_wmma(const _Float16* __restrict__ Qh, const _Float16* __restrict__ Kh,
                     const _Float16* __restrict__ Vh, const int* __restrict__ mask,
                     _Float16* __restrict__ Ctx) {
  __shared__ _Float16 ldsK[2][32 * 64];   // [j][d]
  __shared__ _Float16 ldsV[64 * 32];      // transposed [d][j]
  __shared__ _Float16 ldsP[8 * 16 * 32];  // per-wave P tile [m][j]

  const int tid  = threadIdx.x;
  const int wave = tid >> 5, lane = tid & 31;
  const int l15  = lane & 15;
  const int hi   = (lane >= 16) ? 1 : 0;
  const int b = blockIdx.z, h = blockIdx.y, qb = blockIdx.x;
  const long headBase = (long)(b * H_ + h) * L_ * DPH_;
  const int qr0 = qb * 128 + wave * 16;
  const float NEGINF = -__builtin_inff();

  AFrag aQ[2]; // Q rows as two 16x32 A operands (dph = 64)
  {
    const unsigned int* Qu =
        (const unsigned int*)(Qh + headBase + (long)(qr0 + l15) * DPH_);
    #pragma unroll
    for (int c = 0; c < 2; ++c)
      #pragma unroll
      for (int v = 0; v < 8; ++v) {
        int k = 32 * c + ((v < 4) ? 0 : 16) + (hi ? 8 : 0) + 2 * (v & 3);
        aQ[c].u[v] = Qu[k >> 1];
      }
  }

  float rowmax[8], rowsum[8];
  #pragma unroll
  for (int r = 0; r < 8; ++r) { rowmax[r] = NEGINF; rowsum[r] = 0.f; }
  v8f o[4] = {};

  const unsigned int* ldsVu = (const unsigned int*)ldsV;
  const unsigned int* ldsPu = (const unsigned int*)ldsP;

#if USE_TDM
  if (wave == 0)
    tdm_load_2d(lds_off(&ldsK[0][0]), Kh + headBase, 64, 32, DPH_);
#endif

  for (int kv0 = 0; kv0 < L_; kv0 += 32) {
    const int buf = (kv0 >> 5) & 1;
#if USE_TDM
    if (wave == 0) __builtin_amdgcn_s_wait_tensorcnt(0);  // K[buf] complete
    __syncthreads();          // K[buf] visible; prior V/P reads done
    if (wave == 0 && kv0 + 32 < L_)   // next K tile overlaps this chunk's math
      tdm_load_2d(lds_off(&ldsK[buf ^ 1][0]),
                  Kh + headBase + (long)(kv0 + 32) * DPH_, 64, 32, DPH_);
#else
    __syncthreads();
    { int j = tid >> 3, d0 = (tid & 7) * 8;
      *(uint4*)&ldsK[buf][j * 64 + d0] =
          *(const uint4*)&Kh[headBase + (long)(kv0 + j) * DPH_ + d0]; }
#endif
    { // V transposed: one 16B load + 8 b16 LDS stores per thread
      int j = tid >> 3, d0 = (tid & 7) * 8;
      Q4 vv; vv.v = *(const uint4*)&Vh[headBase + (long)(kv0 + j) * DPH_ + d0];
      #pragma unroll
      for (int i = 0; i < 8; ++i) ldsV[(d0 + i) * 32 + j] = vv.h[i];
      if (kv0 + 64 < L_)   // warm L2 two chunks ahead
        __builtin_prefetch(&Vh[headBase + (long)(kv0 + 64 + j) * DPH_ + d0], 0, 1);
    }
#if !USE_TDM
    __syncthreads();
#endif
    const unsigned int* ldsKu = (const unsigned int*)&ldsK[buf][0];

    // S = Q @ K^T  (16 x 32 as two 16x16 f32 tiles)
    v8f s[2];
    #pragma unroll
    for (int t = 0; t < 2; ++t) {
      v8f sc = {};
      #pragma unroll
      for (int c = 0; c < 2; ++c) {
        AFrag bb; // B[k=d][n=kvcol]: ldsK[(16t+n)][32c + (v<4?0:16) + (hi?8:0) + 2(v&3)]
        int base = (16 * t + l15) * 32 + 16 * c;
        #pragma unroll
        for (int v = 0; v < 8; ++v) bb.u[v] = ldsKu[base + (v & 3) + (hi ? 4 : 0) + ((v < 4) ? 0 : 8)];
        sc = __builtin_amdgcn_wmma_f32_16x16x32_f16(
            false, aQ[c].h, false, bb.h, (short)0, sc, false, false);
      }
      s[t] = sc;
    }

    float madd0 = mask[b * L_ + kv0 + l15]      ? 0.f : NEGINF;
    float madd1 = mask[b * L_ + kv0 + 16 + l15] ? 0.f : NEGINF;
    #pragma unroll
    for (int r = 0; r < 8; ++r) { s[0][r] += madd0; s[1][r] += madd1; }

    // Online softmax per row (row = r + hi*8); stats via half-wave shuffles
    #pragma unroll
    for (int r = 0; r < 8; ++r) {
      float cur = hmax16(fmaxf(s[0][r], s[1][r]));
      float nm  = fmaxf(rowmax[r], cur);
      float nm2 = (nm == NEGINF) ? 0.f : nm;
      float corr = __expf(fminf(rowmax[r] - nm2, 0.f));
      float p0 = __expf(s[0][r] - nm2);
      float p1 = __expf(s[1][r] - nm2);
      float csum = hsum16(p0 + p1);
      rowsum[r] = rowsum[r] * corr + csum;
      rowmax[r] = nm;
      #pragma unroll
      for (int t2 = 0; t2 < 4; ++t2) o[t2][r] *= corr;
      int ml = r + (hi ? 8 : 0);
      ldsP[wave * 512 + ml * 32 + l15]      = (_Float16)p0;
      ldsP[wave * 512 + ml * 32 + 16 + l15] = (_Float16)p1;
    }
    __syncthreads();   // P visible (and V tile complete) before P@V

    // O += P @ V : A = P (16x32), B = V^T tiles (32x16 per 16-dph chunk)
    AFrag aP;
    {
      int rb = wave * 256 + l15 * 16;
      #pragma unroll
      for (int v = 0; v < 8; ++v)
        aP.u[v] = ldsPu[rb + (v & 3) + (hi ? 4 : 0) + ((v < 4) ? 0 : 8)];
    }
    #pragma unroll
    for (int t2 = 0; t2 < 4; ++t2) {
      AFrag bb; // B[k=j][n=d]: ldsV[(16*t2+n)][(v<4?0:16) + (hi?8:0) + 2(v&3)]
      int base = (16 * t2 + l15) * 16;
      #pragma unroll
      for (int v = 0; v < 8; ++v) bb.u[v] = ldsVu[base + (v & 3) + (hi ? 4 : 0) + ((v < 4) ? 0 : 8)];
      o[t2] = __builtin_amdgcn_wmma_f32_16x16x32_f16(
          false, aP.h, false, bb.h, (short)0, o[t2], false, false);
    }
  }

  // Normalize and store context in [b,h,l,d] f16
  #pragma unroll
  for (int r = 0; r < 8; ++r) {
    float inv = (rowsum[r] > 0.f) ? (1.f / rowsum[r]) : 0.f;
    int ml = r + (hi ? 8 : 0);
    #pragma unroll
    for (int t2 = 0; t2 < 4; ++t2)
      Ctx[headBase + (long)(qr0 + ml) * DPH_ + t2 * 16 + l15] =
          (_Float16)(o[t2][r] * inv);
  }
}

extern "C" void kernel_launch(void* const* d_in, const int* in_sizes, int n_in,
                              void* d_out, int out_size, void* d_ws, size_t ws_size,
                              hipStream_t stream) {
  const float* input = (const float*)d_in[0];
  const int*   mask  = (const int*)d_in[1];
  const float* wq = (const float*)d_in[2];
  const float* bq = (const float*)d_in[3];
  const float* wk = (const float*)d_in[4];
  const float* bk = (const float*)d_in[5];
  const float* wv = (const float*)d_in[6];
  const float* bv = (const float*)d_in[7];
  const float* wo = (const float*)d_in[8];
  const float* bo = (const float*)d_in[9];

  char* ws = (char*)d_ws;
  const size_t MB = 1024 * 1024;
  _Float16* Xh  = (_Float16*)(ws + 0);        // 16 MB  input f16 [m][k]
  _Float16* Wqh = (_Float16*)(ws + 16 * MB);
  _Float16* Wkh = (_Float16*)(ws + 18 * MB);
  _Float16* Wvh = (_Float16*)(ws + 20 * MB);
  _Float16* Woh = (_Float16*)(ws + 22 * MB);
  _Float16* Qh  = (_Float16*)(ws + 24 * MB);  // 16 MB [b,h,l,d]
  _Float16* Kh  = (_Float16*)(ws + 40 * MB);  // 16 MB
  _Float16* Vh  = (_Float16*)(ws + 56 * MB);  // 16 MB
  _Float16* Ch  = (_Float16*)(ws + 72 * MB);  // 16 MB context [b,h,l,d]

  cvt_f32_to_f16<<<2048, 256, 0, stream>>>(input, Xh, M_ * D_);
  cvt_f32_to_f16<<<1024, 256, 0, stream>>>(wq, Wqh, D_ * D_);
  cvt_f32_to_f16<<<1024, 256, 0, stream>>>(wk, Wkh, D_ * D_);
  cvt_f32_to_f16<<<1024, 256, 0, stream>>>(wv, Wvh, D_ * D_);
  cvt_f32_to_f16<<<1024, 256, 0, stream>>>(wo, Woh, D_ * D_);

  dim3 gg(M_ / 64, D_ / 128);
  gemm_f16_wmma<0><<<gg, 256, 0, stream>>>(Xh, Wqh, bq, (void*)Qh, 0.125f);
  gemm_f16_wmma<0><<<gg, 256, 0, stream>>>(Xh, Wkh, bk, (void*)Kh, 1.0f);
  gemm_f16_wmma<0><<<gg, 256, 0, stream>>>(Xh, Wvh, bv, (void*)Vh, 1.0f);

  flash_attn_wmma<<<dim3(L_ / 128, H_, B_), 256, 0, stream>>>(Qh, Kh, Vh, mask, Ch);

  gemm_f16_wmma<1><<<gg, 256, 0, stream>>>(Ch, Woh, bo, d_out, 1.0f);
}
